// ConvNet_70720931496461
// MI455X (gfx1250) — matrix operand order repaired and checked by reading, not compiled, not run on hardware
//
#include <hip/hip_runtime.h>

typedef __attribute__((ext_vector_type(16))) _Float16 v16h;
typedef __attribute__((ext_vector_type(8)))  float    v8f;
typedef _Float16 h16;

static __device__ __forceinline__ v8f wmma32(v16h a, v16h b, v8f c) {
  // D = A(16x32 f16) * B(32x16 f16) + C(16x16 f32)
  return __builtin_amdgcn_wmma_f32_16x16x32_f16(
      /*neg_a=*/false, a, /*neg_b=*/false, b,
      /*c_mod=*/(short)0, c, /*reuse_a=*/false, /*reuse_b=*/false);
}

// Async 16-byte global->LDS copy (per-lane), tracked by ASYNCcnt.
static __device__ __forceinline__ void async_copy_b128(const void* gptr, void* lptr) {
  unsigned long long ga = (unsigned long long)(uintptr_t)gptr;
  unsigned int       la = (unsigned int)(uintptr_t)lptr;  // flat[31:0] == LDS offset
  asm volatile("global_load_async_to_lds_b128 %0, %1, off"
               :: "v"(la), "v"(ga) : "memory");
}
static __device__ __forceinline__ void wait_async0() {
  asm volatile("s_wait_asynccnt 0x0" ::: "memory");
}

// A-fragment: contiguous per-lane K runs from an LDS im2col matrix.
static __device__ __forceinline__ v16h a_frag(const h16* rowp_k0, int kb) {
  v16h a;
#pragma unroll
  for (int j = 0; j < 16; ++j) a[j] = rowp_k0[2 * kb + j];
  return a;
}

// B-fragment: contiguous per-lane K run from padded f16 weights [N][ldk].
static __device__ __forceinline__ v16h b_frag(const h16* colp_k0) {
  v16h b;
#pragma unroll
  for (int j = 0; j < 16; ++j) b[j] = colp_k0[j];
  return b;
}

// ---------------------------------------------------------------------------
// Prep: zero LIF/LI states, convert + pad weights to f16 matrices laid out
// [N][K] row-major so B-fragments are contiguous per-lane K runs.
// ---------------------------------------------------------------------------
__global__ void __launch_bounds__(256)
k_init(const float* __restrict__ w1, const float* __restrict__ w2,
       const float* __restrict__ wfc, const float* __restrict__ wout,
       float* __restrict__ states, int stateN,
       h16* __restrict__ w1p, h16* __restrict__ w2p,
       h16* __restrict__ wfcp, h16* __restrict__ woutp) {
  const int tid    = blockIdx.x * blockDim.x + threadIdx.x;
  const int stride = gridDim.x * blockDim.x;
  for (int i = tid; i < stateN; i += stride) states[i] = 0.f;
  for (int i = tid; i < 32 * 32; i += stride) {
    int n = i >> 5, k = i & 31;
    w1p[i] = (h16)((n < 20 && k < 25) ? w1[n * 25 + k] : 0.f);
  }
  for (int i = tid; i < 64 * 512; i += stride) {
    int n = i >> 9, k = i & 511;
    w2p[i] = (h16)((n < 50 && k < 500) ? w2[n * 500 + k] : 0.f);
  }
  for (int i = tid; i < 512 * 800; i += stride) {
    int n = i / 800, k = i - n * 800;
    wfcp[i] = (h16)((n < 500) ? wfc[n * 800 + k] : 0.f);
  }
  for (int i = tid; i < 16 * 512; i += stride) {
    int n = i >> 9, k = i & 511;
    woutp[i] = (h16)((n < 10 && k < 500) ? wout[n * 500 + k] : 0.f);
  }
}

// ---------------------------------------------------------------------------
// Fused conv stage: conv1(GEMM 576x32x32) + LIF + pool  -->  (spikes in LDS)
//                   conv2(GEMM 64x64x512) *10 + LIF + pool --> sp2 global.
// One workgroup (8 waves) per batch image; pooled conv1 spikes never leave
// LDS. Phased 77.7 KB LDS arena (phase-B regions reuse dead phase-A space).
// ---------------------------------------------------------------------------
__global__ void __launch_bounds__(256)
k_conv_stage(const float* __restrict__ xt,
             const h16* __restrict__ w1p, const float* __restrict__ b1,
             float* __restrict__ v0, float* __restrict__ i0,
             const h16* __restrict__ w2p, const float* __restrict__ b2,
             float* __restrict__ v1, float* __restrict__ i1,
             h16* __restrict__ sp2) {
  __shared__ __align__(16) char smem[77696];
  // Phase A layout
  float* imgf = (float*)smem;                  // [0,     3136)  raw f32 image
  h16*   A2a  = (h16*)(smem + 3200);           // [3200,  40064) im2col 576x32
  h16*   sb1  = (h16*)(smem + 40064);          // [40064, 63104) spikes 20x576
  // Phase B layout (overlaps dead phase-A regions)
  h16*   spL  = (h16*)smem;                    // [0,     5760)  pooled 20x144
  h16*   A2b  = (h16*)(smem + 5760);           // [5760,  71296) im2col 64x512
  h16*   sb2  = (h16*)(smem + 71296);          // [71296, 77696) spikes 50x64

  const int b    = blockIdx.x;
  const int tid  = threadIdx.x;
  const int lane = tid & 31;
  const int wid  = tid >> 5;
  const int kb   = (lane >> 4) * 8;

  // --- Phase A: conv1 ------------------------------------------------------
  // Async stage raw image: 3136 B = 196 x 16-byte chunks (ASYNCcnt path).
  for (int c = tid; c < 196; c += 256)
    async_copy_b128(xt + b * 784 + c * 4, imgf + c * 4);
  wait_async0();
  __syncthreads();

  // Build im2col (f32 -> f16): A2a[row][k], k = ky*5+kx (pad->0).
  for (int p = tid; p < 576 * 32; p += 256) {
    const int row = p >> 5, k = p & 31;
    h16 v = (h16)0.f;
    if (k < 25) {
      const int oy = row / 24, ox = row - oy * 24;
      const int ky = k / 5, kx = k - ky * 5;
      v = (h16)imgf[(oy + ky) * 28 + (ox + kx)];
    }
    A2a[p] = v;
  }
  __syncthreads();

  for (int mt = wid; mt < 36; mt += 8) {           // wave-uniform tiling
    const v16h a = a_frag(A2a + (mt * 16 + (lane & 15)) * 32, kb);
#pragma unroll
    for (int nt = 0; nt < 2; ++nt) {
      const v16h bf = b_frag(w1p + (nt * 16 + (lane & 15)) * 32 + (lane >> 4) * 16);
      v8f acc = {};
      acc = wmma32(a, bf, acc);
      const int n = nt * 16 + (lane & 15);
      if (n < 20) {
        const float bias = b1[n];
#pragma unroll
        for (int r = 0; r < 8; ++r) {
          const int m   = mt * 16 + (lane >> 4) * 8 + r;
          const int idx = (b * 20 + n) * 576 + m;
          const float inp = acc[r] + bias;
          const float v   = v0[idx], cur = i0[idx];
          const float vd  = 0.9f * v + 0.1f * cur;     // v + dt*tau_mem*(-v+i)
          const float z   = (vd > 1.0f) ? 1.f : 0.f;    // spike
          v0[idx] = (1.f - z) * vd;
          i0[idx] = 0.8f * cur + inp;                   // i*(1-dt*tau_syn)+inp
          sb1[n * 576 + m] = (h16)z;
        }
      }
    }
  }
  __syncthreads();

  // 2x2 maxpool of conv1 spikes -> spL (stays in LDS).
  for (int p = tid; p < 20 * 144; p += 256) {
    const int n = p / 144, rem = p - n * 144;
    const int py = rem / 12, px = rem - py * 12;
    const h16* s = sb1 + n * 576 + (2 * py) * 24 + 2 * px;
    float m = (float)s[0];
    m = fmaxf(m, (float)s[1]);
    m = fmaxf(m, (float)s[24]);
    m = fmaxf(m, (float)s[25]);
    spL[p] = (h16)m;
  }
  __syncthreads();

  // --- Phase B: conv2 ------------------------------------------------------
  // Build im2col: k = c*25 + ky*5 + kx, row = oy*8+ox (pad->0).
  for (int p = tid; p < 64 * 512; p += 256) {
    const int row = p >> 9, k = p & 511;
    h16 v = (h16)0.f;
    if (k < 500) {
      const int c  = k / 25, rem = k - c * 25;
      const int ky = rem / 5, kx = rem - ky * 5;
      const int oy = row >> 3, ox = row & 7;
      v = spL[c * 144 + (oy + ky) * 12 + (ox + kx)];
    }
    A2b[p] = v;
  }
  __syncthreads();

  for (int tt = wid; tt < 16; tt += 8) {           // 4 m-tiles x 4 n-tiles
    const int mt = tt >> 2, nt = tt & 3;
    const h16* arow = A2b + (mt * 16 + (lane & 15)) * 512;
    const h16* wcol = w2p + (nt * 16 + (lane & 15)) * 512 + (lane >> 4) * 16;
    v8f acc = {};
    for (int ks = 0; ks < 16; ++ks) {
      const v16h a  = a_frag(arow + ks * 32, kb);
      const v16h bf = b_frag(wcol + ks * 32);
      acc = wmma32(a, bf, acc);
    }
    const int n = nt * 16 + (lane & 15);
    if (n < 50) {
      const float bias = b2[n];
#pragma unroll
      for (int r = 0; r < 8; ++r) {
        const int m   = mt * 16 + (lane >> 4) * 8 + r;
        const int idx = (b * 50 + n) * 64 + m;
        const float inp = 10.f * (acc[r] + bias);
        const float v   = v1[idx], cur = i1[idx];
        const float vd  = 0.9f * v + 0.1f * cur;
        const float z   = (vd > 1.0f) ? 1.f : 0.f;
        v1[idx] = (1.f - z) * vd;
        i1[idx] = 0.8f * cur + inp;
        sb2[n * 64 + m] = (h16)z;
      }
    }
  }
  __syncthreads();

  for (int p = tid; p < 50 * 16; p += 256) {       // pool -> [B,50,4,4] flat
    const int n = p >> 4, rem = p & 15;
    const int py = rem >> 2, px = rem & 3;
    const h16* s = sb2 + n * 64 + (2 * py) * 8 + 2 * px;
    float m = (float)s[0];
    m = fmaxf(m, (float)s[1]);
    m = fmaxf(m, (float)s[8]);
    m = fmaxf(m, (float)s[9]);
    sp2[b * 800 + p] = (h16)m;
  }
}

// ---------------------------------------------------------------------------
// Kernel 2: FC 800->500 GEMM + LIF, then fused 500->10 readout GEMM + LI.
// One workgroup per 16 batch rows; A tile staged async (contiguous 25.6 KB).
// ---------------------------------------------------------------------------
__global__ void __launch_bounds__(256)
k_fc_lif_out(const h16* __restrict__ sp2, const h16* __restrict__ wfcp,
             const float* __restrict__ bfc, const h16* __restrict__ woutp,
             float* __restrict__ v2, float* __restrict__ i2,
             float* __restrict__ vo, float* __restrict__ io,
             float* __restrict__ out_t) {
  __shared__ __align__(16) h16 atile[16 * 800];   // == sp2[b0*800 ..), flat
  __shared__ __align__(16) h16 zl[16 * 512];
  const int b0   = blockIdx.x * 16;
  const int tid  = threadIdx.x;
  const int lane = tid & 31;
  const int wid  = tid >> 5;
  const int kb   = (lane >> 4) * 8;

  // Async stage: 1600 x 16-byte chunks of the contiguous activation slice.
  for (int c = tid; c < 1600; c += 256)
    async_copy_b128(sp2 + b0 * 800 + c * 8, atile + c * 8);
  wait_async0();
  __syncthreads();

  for (int nt = wid * 4; nt < wid * 4 + 4; ++nt) { // 32 n-tiles over 8 waves
    const h16* arow = atile + (lane & 15) * 800;
    const h16* wcol = wfcp + (nt * 16 + (lane & 15)) * 800 + (lane >> 4) * 16;
    v8f acc = {};
    for (int ks = 0; ks < 25; ++ks) {              // K = 800 = 25*32 exact
      __builtin_prefetch(wcol + ks * 32 + 32, 0, 3);
      const v16h a  = a_frag(arow + ks * 32, kb);
      const v16h bf = b_frag(wcol + ks * 32);
      acc = wmma32(a, bf, acc);
    }
    const int n = nt * 16 + (lane & 15);
    const float bias = (n < 500) ? bfc[n] : 0.f;
#pragma unroll
    for (int r = 0; r < 8; ++r) {
      const int m = (lane >> 4) * 8 + r;           // local batch row 0..15
      h16 zspk = (h16)0.f;
      if (n < 500) {
        const int idx = (b0 + m) * 500 + n;
        const float inp = acc[r] + bias;
        const float v   = v2[idx], cur = i2[idx];
        const float vd  = 0.9f * v + 0.1f * cur;
        const float z   = (vd > 1.0f) ? 1.f : 0.f;
        v2[idx] = (1.f - z) * vd;
        i2[idx] = 0.8f * cur + inp;
        zspk = (h16)z;                             // relu(z) == z for spikes
      }
      zl[m * 512 + n] = zspk;
    }
  }
  __syncthreads();

  if (wid == 0) {                                  // readout GEMM + LI cell
    const h16* arow = zl + (lane & 15) * 512;
    const h16* wcol = woutp + (lane & 15) * 512 + (lane >> 4) * 16;
    v8f acc = {};
    for (int ks = 0; ks < 16; ++ks) {
      const v16h a  = a_frag(arow + ks * 32, kb);
      const v16h bf = b_frag(wcol + ks * 32);
      acc = wmma32(a, bf, acc);
    }
    const int n = lane & 15;
    if (n < 10) {
#pragma unroll
      for (int r = 0; r < 8; ++r) {
        const int m   = (lane >> 4) * 8 + r;
        const int idx = (b0 + m) * 10 + n;
        const float v = vo[idx], cur = io[idx];
        const float vn = 0.9f * v + 0.1f * cur;    // LI voltage (output)
        vo[idx] = vn;
        io[idx] = 0.8f * cur + acc[r];
        out_t[(b0 + m) * 10 + n] = vn;
      }
    }
  }
}

// ---------------------------------------------------------------------------
// Host launcher: workspace carve + T=64 sequential steps (graph-capture safe).
// ---------------------------------------------------------------------------
extern "C" void kernel_launch(void* const* d_in, const int* in_sizes, int n_in,
                              void* d_out, int out_size, void* d_ws, size_t ws_size,
                              hipStream_t stream) {
  (void)in_sizes; (void)n_in; (void)out_size; (void)ws_size;
  const float* x    = (const float*)d_in[0];
  const float* w1   = (const float*)d_in[1];
  const float* b1   = (const float*)d_in[2];
  const float* w2   = (const float*)d_in[3];
  const float* b2   = (const float*)d_in[4];
  const float* wfc  = (const float*)d_in[5];
  const float* bfc  = (const float*)d_in[6];
  const float* wout = (const float*)d_in[7];
  float* out = (float*)d_out;

  const size_t V0N = 256 * 20 * 576;   // 2,949,120
  const size_t V1N = 256 * 50 * 64;    // 819,200
  const size_t V2N = 256 * 500;        // 128,000
  const size_t VON = 256 * 10;         // 2,560
  const int stateN = (int)(2 * (V0N + V1N + V2N + VON));

  float* states = (float*)d_ws;
  float* v0 = states;       float* i0 = v0 + V0N;
  float* v1 = i0 + V0N;     float* i1 = v1 + V1N;
  float* v2 = i1 + V1N;     float* i2 = v2 + V2N;
  float* vo = i2 + V2N;     float* io = vo + VON;
  h16* sp2   = (h16*)(io + VON);            // [B,800] spikes (16B-aligned)
  h16* w1p   = sp2 + 256 * 800;             // [32][32]
  h16* w2p   = w1p + 32 * 32;               // [64][512]
  h16* wfcp  = w2p + 64 * 512;              // [512][800]
  h16* woutp = wfcp + 512 * 800;            // [16][512]

  hipLaunchKernelGGL(k_init, dim3(512), dim3(256), 0, stream,
                     w1, w2, wfc, wout, states, stateN, w1p, w2p, wfcp, woutp);

  for (int t = 0; t < 64; ++t) {
    hipLaunchKernelGGL(k_conv_stage, dim3(256), dim3(256), 0, stream,
                       x + (size_t)t * 256 * 784, w1p, b1, v0, i0,
                       w2p, b2, v1, i1, sp2);
    hipLaunchKernelGGL(k_fc_lif_out, dim3(16), dim3(256), 0, stream,
                       sp2, wfcp, bfc, woutp, v2, i2, vo, io,
                       out + (size_t)t * 2560);
  }
}
